// PNA_60292750901470
// MI455X (gfx1250) — matrix operand order repaired and checked by reading, not compiled, not run on hardware
//
#include <hip/hip_runtime.h>
#include <math.h>

// -------- CDNA5 vector types --------
typedef __attribute__((ext_vector_type(16))) _Float16 v16h;
typedef __attribute__((ext_vector_type(8)))  _Float16 v8h;
typedef __attribute__((ext_vector_type(8)))  float    v8f;

#define D_NODE 128
#define EMBED  64
#define KTOT   (D_NODE + 12 * EMBED)   // 896
#define FMAXV  3.402823466e+38f

#define WMMA16(a, b, c) \
  __builtin_amdgcn_wmma_f32_16x16x32_f16(false, (a), false, (b), (short)0, (c), false, false)

__device__ __forceinline__ float mishf(float x) {
  float sp = (x > 20.0f) ? x : log1pf(__expf(x));
  return x * tanhf(sp);
}

// jnp.nan_to_num semantics: NaN->0, +inf->FLT_MAX, -inf->-FLT_MAX
__device__ __forceinline__ float nan2num(float x) {
  if (x != x) return 0.0f;
  if (x > FMAXV) return FMAXV;
  if (x < -FMAXV) return -FMAXV;
  return x;
}

// float atomic max/min via sign-aware integer ordering (compiles everywhere)
__device__ __forceinline__ void atomicMaxF(float* addr, float v) {
  if (v >= 0.0f) atomicMax((int*)addr, __float_as_int(v));
  else           atomicMin((unsigned int*)addr, __float_as_uint(v));
}
__device__ __forceinline__ void atomicMinF(float* addr, float v) {
  if (v >= 0.0f) atomicMin((int*)addr, __float_as_int(v));
  else           atomicMax((unsigned int*)addr, __float_as_uint(v));
}

// -------- prep kernels --------
__global__ void pna_init_kernel(float* __restrict__ sumb, float* __restrict__ sumq,
                                float* __restrict__ mxb, float* __restrict__ mnb,
                                float* __restrict__ counts, float* __restrict__ ls_sum,
                                int total, int N) {
  int i = blockIdx.x * blockDim.x + threadIdx.x;
  if (i < total) {
    sumb[i] = 0.0f;
    sumq[i] = 0.0f;
    mxb[i]  = -FMAXV;
    mnb[i]  =  FMAXV;
  }
  if (i < N) counts[i] = 0.0f;
  if (i == 0) *ls_sum = 0.0f;
}

__global__ void pna_cvt_nodes_kernel(const float* __restrict__ nodes,
                                     _Float16* __restrict__ nodes_h, int total) {
  int i = blockIdx.x * blockDim.x + threadIdx.x;
  if (i < total) nodes_h[i] = (_Float16)nodes[i];
}

// Transpose + convert weights: Wt[n][k] = W[k][n], f32 -> f16
__global__ void pna_cvt_weights_kernel(const float* __restrict__ W_edge,
                                       const float* __restrict__ W_node,
                                       _Float16* __restrict__ WtE,
                                       _Float16* __restrict__ WtN) {
  int i = blockIdx.x * blockDim.x + threadIdx.x;
  if (i < EMBED * (2 * D_NODE)) {               // 64*256
    int n = i / (2 * D_NODE), k = i % (2 * D_NODE);
    WtE[i] = (_Float16)W_edge[k * EMBED + n];
  }
  if (i < EMBED * KTOT) {                       // 64*896
    int n = i / KTOT, k = i % KTOT;
    WtN[i] = (_Float16)W_node[k * EMBED + n];
  }
}

// -------- edge stage: gather -> WMMA (16x256 @ 256x64) -> mish -> atomic multi-aggregate --------
__global__ __launch_bounds__(256) void pna_edge_kernel(
    const _Float16* __restrict__ nodes_h,
    const int* __restrict__ senders,
    const int* __restrict__ receivers,
    const _Float16* __restrict__ WtE,      // [64][256] f16
    const float* __restrict__ b_edge,
    float* __restrict__ sumb, float* __restrict__ sumq,
    float* __restrict__ mxb,  float* __restrict__ mnb,
    float* __restrict__ counts, int E) {
  const int lane    = threadIdx.x & 31;
  const int wave    = threadIdx.x >> 5;
  const int lidx    = lane & 15;
  const int halfsel = lane >> 4;
  const int tile    = blockIdx.x * 8 + wave;
  const int ebase   = tile * 16;
  if (ebase >= E) return;

  // A-fragment row for this lane: edge ebase + lidx
  const int myEdge = ebase + lidx;
  const int s = senders[myEdge];
  const int r = receivers[myEdge];
  const _Float16* srow = nodes_h + (size_t)s * D_NODE;
  const _Float16* rrow = nodes_h + (size_t)r * D_NODE;

  v8f acc[4] = {};
  #pragma unroll
  for (int kc = 0; kc < 8; ++kc) {
    // 16-bit A layout: lane half holds contiguous K run [kc*32 + hs*16, +16) as two 8-half chunks
    const int kA0 = kc * 32 + halfsel * 16;
    const int kA1 = kA0 + 8;
    // K < 128 -> sender features, K >= 128 -> receiver features (uniform per kc)
    const _Float16* arow = (kc < 4) ? srow : (rrow - D_NODE);
    v8h lo = *(const v8h*)(arow + kA0);
    v8h hi = *(const v8h*)(arow + kA1);
    v16h a = __builtin_shufflevector(lo, hi, 0,1,2,3,4,5,6,7,8,9,10,11,12,13,14,15);
    // 16-bit B layout: lane holds contiguous K run [kc*32 + hs*16, +16) of column lidx
    const int kB = kc * 32 + halfsel * 16;
    #pragma unroll
    for (int nt = 0; nt < 4; ++nt) {
      v16h b = *(const v16h*)(WtE + (size_t)(nt * 16 + lidx) * (2 * D_NODE) + kB);
      acc[nt] = WMMA16(a, b, acc[nt]);
    }
  }

  // Receivers of the 8 C-rows this lane owns (M = v + hs*8)
  int recvm[8];
  #pragma unroll
  for (int v = 0; v < 8; ++v) recvm[v] = receivers[ebase + v + halfsel * 8];

  #pragma unroll
  for (int nt = 0; nt < 4; ++nt) {
    const int n = nt * 16 + lidx;
    const float bias = b_edge[n];
    #pragma unroll
    for (int v = 0; v < 8; ++v) {
      float e = mishf(acc[nt][v] + bias);
      float* base = (float*)0;
      size_t off = (size_t)recvm[v] * EMBED + n;
      (void)base;
      atomicAdd(sumb + off, e);
      atomicAdd(sumq + off, e * e);
      atomicMaxF(mxb + off, e);
      atomicMinF(mnb + off, e);
    }
  }
  if (lane < 16) atomicAdd(&counts[r], 1.0f);   // one count per edge
}

// -------- sum of log1p(degree) for scaler normalization --------
__global__ void pna_degree_reduce_kernel(const float* __restrict__ counts,
                                         float* __restrict__ ls_sum, int N) {
  int i = blockIdx.x * blockDim.x + threadIdx.x;
  float v = (i < N) ? log1pf(counts[i]) : 0.0f;
  #pragma unroll
  for (int off = 16; off > 0; off >>= 1) v += __shfl_down(v, off, 32);
  if ((threadIdx.x & 31) == 0) atomicAdd(ls_sum, v);
}

// -------- node stage: build 896-wide input on the fly -> WMMA -> mish --------
__global__ __launch_bounds__(256) void pna_node_kernel(
    const _Float16* __restrict__ nodes_h,
    const _Float16* __restrict__ WtN,      // [64][896] f16
    const float* __restrict__ b_node,
    const float* __restrict__ sumb, const float* __restrict__ sumq,
    const float* __restrict__ mxb,  const float* __restrict__ mnb,
    const float* __restrict__ counts, const float* __restrict__ ls_sum,
    float* __restrict__ out, int N) {
  const int lane    = threadIdx.x & 31;
  const int wave    = threadIdx.x >> 5;
  const int lidx    = lane & 15;
  const int halfsel = lane >> 4;
  const int tile    = blockIdx.x * 8 + wave;
  const int nbase   = tile * 16;
  if (nbase >= N) return;

  const float ls_mean = *ls_sum / (float)N;
  const int row = min(nbase + lidx, N - 1);
  const float cntRaw = counts[row];
  const float cnt    = fmaxf(cntRaw, 1.0f);
  const float rcnt   = 1.0f / cnt;
  const float ls     = log1pf(cntRaw) / ls_mean;
  const float inv_ls = 1.0f / ls;                 // +inf when degree==0 (matches ls**-1)
  const float* srow = sumb + (size_t)row * EMBED;
  const float* qrow = sumq + (size_t)row * EMBED;
  const float* xrow = mxb  + (size_t)row * EMBED;
  const float* nrow = mnb  + (size_t)row * EMBED;

  v8f acc[4] = {};
  #pragma unroll
  for (int kc = 0; kc < 28; ++kc) {
    const int kA0 = kc * 32 + halfsel * 16;
    const int kA1 = kA0 + 8;
    v16h a;
    if (kc < 4) {
      const _Float16* np = nodes_h + (size_t)row * D_NODE;
      v8h lo = *(const v8h*)(np + kA0);
      v8h hi = *(const v8h*)(np + kA1);
      a = __builtin_shufflevector(lo, hi, 0,1,2,3,4,5,6,7,8,9,10,11,12,13,14,15);
    } else {
      #pragma unroll
      for (int h = 0; h < 2; ++h) {
        const int idx  = ((h == 0) ? kA0 : kA1) - D_NODE;  // 0..767, 8-aligned
        const int sc   = idx >> 8;                          // scaler: 1, ls, 1/ls
        const int comp = (idx >> 6) & 3;                    // mean/std/max/min
        const int n0   = idx & 63;
        const float scl = (sc == 0) ? 1.0f : ((sc == 1) ? ls : inv_ls);
        #pragma unroll
        for (int t = 0; t < 8; ++t) {
          const int n = n0 + t;
          float v;
          if (comp == 0) {
            v = srow[n] * rcnt;                             // mean
          } else if (comp == 1) {                           // std (single-pass var)
            float m = srow[n] * rcnt;
            v = sqrtf(fmaxf(qrow[n] * rcnt - m * m, 0.0f) + 1e-10f);
          } else if (comp == 2) {
            v = xrow[n];                                    // max
          } else {
            v = nrow[n];                                    // min
          }
          a[h * 8 + t] = (_Float16)nan2num(scl * v);
        }
      }
    }
    const int kB = kc * 32 + halfsel * 16;
    #pragma unroll
    for (int nt = 0; nt < 4; ++nt) {
      v16h b = *(const v16h*)(WtN + (size_t)(nt * 16 + lidx) * KTOT + kB);
      acc[nt] = WMMA16(a, b, acc[nt]);
    }
  }

  #pragma unroll
  for (int nt = 0; nt < 4; ++nt) {
    const int n = nt * 16 + lidx;
    const float bias = b_node[n];
    #pragma unroll
    for (int v = 0; v < 8; ++v) {
      const int node = nbase + v + halfsel * 8;
      if (node < N) out[(size_t)node * EMBED + n] = mishf(acc[nt][v] + bias);
    }
  }
}

extern "C" void kernel_launch(void* const* d_in, const int* in_sizes, int n_in,
                              void* d_out, int out_size, void* d_ws, size_t ws_size,
                              hipStream_t stream) {
  const float* nodes    = (const float*)d_in[0];
  const int*   senders  = (const int*)d_in[1];
  const int*   receivers= (const int*)d_in[2];
  const float* W_edge   = (const float*)d_in[3];
  const float* b_edge   = (const float*)d_in[4];
  const float* W_node   = (const float*)d_in[5];
  const float* b_node   = (const float*)d_in[6];
  float* out = (float*)d_out;

  const int N = in_sizes[0] / D_NODE;   // 50000
  const int E = in_sizes[1];            // 800000

  // ---- carve workspace (≈65 MB) ----
  size_t off = 0;
  auto take = [&](size_t bytes) -> void* {
    void* p = (char*)d_ws + off;
    off += (bytes + 255) & ~(size_t)255;
    return p;
  };
  _Float16* nodes_h = (_Float16*)take((size_t)N * D_NODE * sizeof(_Float16));
  _Float16* WtE     = (_Float16*)take((size_t)EMBED * 2 * D_NODE * sizeof(_Float16));
  _Float16* WtN     = (_Float16*)take((size_t)EMBED * KTOT * sizeof(_Float16));
  float* counts = (float*)take((size_t)N * sizeof(float));
  float* sumb   = (float*)take((size_t)N * EMBED * sizeof(float));
  float* sumq   = (float*)take((size_t)N * EMBED * sizeof(float));
  float* mxb    = (float*)take((size_t)N * EMBED * sizeof(float));
  float* mnb    = (float*)take((size_t)N * EMBED * sizeof(float));
  float* ls_sum = (float*)take(sizeof(float));

  const int B = 256;
  // init accumulators (runs every call -> deterministic across graph replays)
  {
    int total = N * EMBED;
    pna_init_kernel<<<(total + B - 1) / B, B, 0, stream>>>(sumb, sumq, mxb, mnb,
                                                           counts, ls_sum, total, N);
  }
  // f32 -> f16 node features
  {
    int total = N * D_NODE;
    pna_cvt_nodes_kernel<<<(total + B - 1) / B, B, 0, stream>>>(nodes, nodes_h, total);
  }
  // transpose+convert weights
  {
    int total = EMBED * KTOT;
    pna_cvt_weights_kernel<<<(total + B - 1) / B, B, 0, stream>>>(W_edge, W_node, WtE, WtN);
  }
  // edge MLP + multi-aggregator scatter
  {
    int tiles = (E + 15) / 16;
    pna_edge_kernel<<<(tiles + 7) / 8, B, 0, stream>>>(nodes_h, senders, receivers, WtE,
                                                       b_edge, sumb, sumq, mxb, mnb,
                                                       counts, E);
  }
  // degree scaler normalization
  pna_degree_reduce_kernel<<<(N + B - 1) / B, B, 0, stream>>>(counts, ls_sum, N);
  // node MLP
  {
    int tiles = (N + 15) / 16;
    pna_node_kernel<<<(tiles + 7) / 8, B, 0, stream>>>(nodes_h, WtN, b_node, sumb, sumq,
                                                       mxb, mnb, counts, ls_sum, out, N);
  }
  (void)n_in; (void)out_size; (void)ws_size;
}
